// ConcatModelMultiHead_7636451852445
// MI455X (gfx1250) — matrix operand order repaired and checked by reading, not compiled, not run on hardware
//
#include <hip/hip_runtime.h>
#include <hip/hip_bf16.h>

#define N_NODES  100000
#define N_EDGES  600000
#define N_GRAPHS 256
#define HID      128
#define IN_DIM   7
#define BN_EPS   1e-5f
#define PITCH    136   // LDS row pitch in bf16 elems: 272B = 17*16B -> conflict-friendly, 16B aligned

typedef __attribute__((ext_vector_type(16))) __bf16 v16bf;
typedef __attribute__((ext_vector_type(8)))  __bf16 v8bf;
typedef __attribute__((ext_vector_type(8)))  float  v8f;

#if defined(__AMDGCN__) && \
    __has_builtin(__builtin_amdgcn_global_load_async_to_lds_b128) && \
    __has_builtin(__builtin_amdgcn_s_wait_asynccnt)
#define HAVE_ASYNC_LDS 1
typedef int v4i __attribute__((vector_size(16)));
typedef __attribute__((address_space(1))) v4i* as1_v4i_p;   // global src
typedef __attribute__((address_space(3))) v4i* as3_v4i_p;   // LDS dst
#else
#define HAVE_ASYNC_LDS 0
#endif

__device__ inline __bf16 f2bf(float f) {
  unsigned u = __builtin_bit_cast(unsigned, f);
  unsigned r = u + 0x7FFFu + ((u >> 16) & 1u);   // round-to-nearest-even
  unsigned short h = (unsigned short)(r >> 16);
  return __builtin_bit_cast(__bf16, h);
}

__device__ inline float bf2f(__bf16 b) {
  unsigned u = ((unsigned)__builtin_bit_cast(unsigned short, b)) << 16;
  return __builtin_bit_cast(float, u);
}

__device__ inline v16bf cat16(v8bf a, v8bf b) {
  return __builtin_shufflevector(a, b, 0, 1, 2, 3, 4, 5, 6, 7,
                                        8, 9, 10, 11, 12, 13, 14, 15);
}

__device__ inline void atomicAddF(float* p, float v) {
  __hip_atomic_fetch_add(p, v, __ATOMIC_RELAXED, __HIP_MEMORY_SCOPE_AGENT);
}

// ---------------- utility kernels ----------------
__global__ void zero_f(float* __restrict__ p, int n) {
  int i = blockIdx.x * blockDim.x + threadIdx.x;
  if (i < n) p[i] = 0.f;
}

__global__ void deg_count(const int* __restrict__ ei, float* __restrict__ deg, int ne) {
  int e = blockIdx.x * blockDim.x + threadIdx.x;
  if (e < ne) atomicAddF(&deg[ei[ne + e]], 1.f);   // dst row of edge_index
}

__global__ void make_dinv(float* __restrict__ deg, int n) {
  int i = blockIdx.x * blockDim.x + threadIdx.x;
  if (i < n) deg[i] = rsqrtf(deg[i] + 1.f);        // +1 self loop
}

// W[k][n] (row-major KxN) -> Wt[n][k] bf16, so B fragments are contiguous in K
__global__ void cvt_bf16_t(const float* __restrict__ w, __bf16* __restrict__ o, int n) {
  int i = blockIdx.x * blockDim.x + threadIdx.x;
  if (i >= n) return;
  int k = i >> 7, c = i & (HID - 1);
  o[c * HID + k] = f2bf(w[i]);
}

// ---------------- layer-0 GEMM: [N,7] @ [7,128] (trivial K, scalar) ----------------
__global__ void gemm_in7(const float* __restrict__ x, const float* __restrict__ W,
                         float* __restrict__ out, int n) {
  int i = blockIdx.x * blockDim.x + threadIdx.x;
  if (i >= n * HID) return;
  int node = i >> 7, c = i & (HID - 1);
  float acc = 0.f;
#pragma unroll
  for (int k = 0; k < IN_DIM; ++k)
    acc = fmaf(x[node * IN_DIM + k], W[k * HID + c], acc);
  out[i] = acc;
}

// ---------------- WMMA GEMM: bf16 [N,128] @ bf16 [128,128] -> f32 ----------------
// 256 threads = 8 waves. Block covers 128 rows; wave w owns rows [16w,16w+16) and
// sweeps all 8 column tiles (32 WMMAs/wave). A/W staged in LDS; async DMA when available.
__global__ __launch_bounds__(256) void gemm128_wmma(
    const __bf16* __restrict__ hbf, const __bf16* __restrict__ Wt,
    float* __restrict__ out, int n) {
  __shared__ __bf16 sW[HID * PITCH];   // transposed W: sW[nrow][k]
  __shared__ __bf16 sA[HID * PITCH];   // 128 rows of activations
  const int t = threadIdx.x;
  const int mblk = blockIdx.x * 128;
  const bool full = (mblk + 128 <= n);

#if HAVE_ASYNC_LDS
  if (full) {
    // 16-byte chunks: W = 1024 chunks, A = 2048 chunks; per-lane async DMA to LDS
    for (int i16 = t; i16 < 1024; i16 += 256) {
      int r = i16 >> 4, c8 = (i16 & 15) * 8;
      __builtin_amdgcn_global_load_async_to_lds_b128(
          (as1_v4i_p)(size_t)(Wt + r * HID + c8),
          (as3_v4i_p)(unsigned)(size_t)&sW[r * PITCH + c8], 0, 0);
    }
    for (int i16 = t; i16 < 2048; i16 += 256) {
      int r = i16 >> 4, c8 = (i16 & 15) * 8;
      __builtin_amdgcn_global_load_async_to_lds_b128(
          (as1_v4i_p)(size_t)(hbf + (size_t)(mblk + r) * HID + c8),
          (as3_v4i_p)(unsigned)(size_t)&sA[r * PITCH + c8], 0, 0);
    }
    __builtin_amdgcn_s_wait_asynccnt(0);
  } else
#endif
  {
    const v8bf vz = {};
    for (int i16 = t; i16 < 1024; i16 += 256) {
      int r = i16 >> 4, c8 = (i16 & 15) * 8;
      *(v8bf*)&sW[r * PITCH + c8] = *(const v8bf*)(Wt + r * HID + c8);
    }
    for (int i16 = t; i16 < 2048; i16 += 256) {
      int r = i16 >> 4, c8 = (i16 & 15) * 8;
      int row = mblk + r;
      *(v8bf*)&sA[r * PITCH + c8] =
          (row < n) ? *(const v8bf*)(hbf + (size_t)row * HID + c8) : vz;
    }
  }
  __syncthreads();

  const int lane = t & 31;
  const int wv   = t >> 5;
  const int hi   = lane >> 4;           // half-wave select per ISA layout
  const int lo   = lane & 15;
  const int mstrip = wv * 16;

  // A 16x32 bf16 frags: lanes 0-15 -> K runs {0..7,16..23}; lanes 16-31 -> {8..15,24..31}
  v16bf afr[4];
#pragma unroll
  for (int kc = 0; kc < 4; ++kc) {
    const __bf16* ap = &sA[(mstrip + lo) * PITCH + kc * 32 + hi * 8];
    afr[kc] = cat16(*(const v8bf*)ap, *(const v8bf*)(ap + 16));
  }

#pragma unroll
  for (int nt = 0; nt < 8; ++nt) {
    const int nn = nt * 16 + lo;
    v8f acc = {};
#pragma unroll
    for (int kc = 0; kc < 4; ++kc) {
      // B 32x16 bf16: column nn; lanes 0-15 -> K runs {0..7,16..23}; lanes 16-31 -> {8..15,24..31}
      const __bf16* bp = &sW[nn * PITCH + kc * 32 + hi * 8];
      v16bf b = cat16(*(const v8bf*)bp, *(const v8bf*)(bp + 16));
      acc = __builtin_amdgcn_wmma_f32_16x16x32_bf16(false, afr[kc], false, b,
                                                    (short)0, acc, false, false);
    }
    // C/D: VGPR r -> row mblk + mstrip + r + 8*hi, col nn
    const int mbase = mblk + mstrip + hi * 8;
    if (full) {                         // block-uniform: plain stores, imm offsets
      float* op = out + (size_t)mbase * HID + nn;
#pragma unroll
      for (int r = 0; r < 8; ++r) op[(size_t)r * HID] = acc[r];
    } else {
#pragma unroll
      for (int r = 0; r < 8; ++r) {
        int m = mbase + r;
        if (m < n) out[(size_t)m * HID + nn] = acc[r];
      }
    }
  }
}

// ---------------- agg init: self-loop term + bias ----------------
__global__ void init_agg(const float* __restrict__ hl, const float* __restrict__ dinv,
                         const float* __restrict__ b, float* __restrict__ agg, int n) {
  int i = blockIdx.x * blockDim.x + threadIdx.x;
  if (i >= n * HID) return;
  int node = i >> 7, c = i & (HID - 1);
  float d = dinv[node];
  agg[i] = fmaf(hl[i], d * d, b[c]);
}

// ---------------- edge scatter: agg[dst] += hl[src] * dinv[src]*dinv[dst] ----------------
__global__ void edge_scatter(const int* __restrict__ ei, const float* __restrict__ hl,
                             const float* __restrict__ dinv, float* __restrict__ agg, int ne) {
  unsigned tid = blockIdx.x * blockDim.x + threadIdx.x;
  int e  = tid >> 5;
  int c4 = (tid & 31) * 4;
  if (e >= ne) return;
  int s = ei[e], d = ei[ne + e];
  float nrm = dinv[s] * dinv[d];
  const float4 v = *(const float4*)(hl + (size_t)s * HID + c4);
  float* ap = agg + (size_t)d * HID + c4;
  atomicAddF(ap + 0, v.x * nrm);
  atomicAddF(ap + 1, v.y * nrm);
  atomicAddF(ap + 2, v.z * nrm);
  atomicAddF(ap + 3, v.w * nrm);
}

// ---------------- BatchNorm(eval) + ReLU -> bf16 activations ----------------
__global__ void bn_relu(const float* __restrict__ agg,
                        const float* __restrict__ g, const float* __restrict__ be,
                        const float* __restrict__ mu, const float* __restrict__ va,
                        int layer, __bf16* __restrict__ h, int n) {
  int i = blockIdx.x * blockDim.x + threadIdx.x;
  if (i >= n * HID) return;
  int c = (i & (HID - 1)) + layer * HID;
  float v = (agg[i] - mu[c]) * (g[c] * rsqrtf(va[c] + BN_EPS)) + be[c];
  h[i] = f2bf(fmaxf(v, 0.f));
}

// ---------------- mean pool (atomic sums + counts) ----------------
__global__ void pool(const __bf16* __restrict__ h, const int* __restrict__ batch,
                     float* __restrict__ sums, float* __restrict__ cnt, int n) {
  int i = blockIdx.x * blockDim.x + threadIdx.x;
  if (i >= n * HID) return;
  int node = i >> 7, c = i & (HID - 1);
  int g = batch[node];
  atomicAddF(&sums[g * HID + c], bf2f(h[i]));
  if (c == 0) atomicAddF(&cnt[g], 1.f);
}

// ---------------- head: pocket MLP + concat + classifier, 1 block / graph ----------------
__global__ __launch_bounds__(128) void head(
    const float* __restrict__ sums, const float* __restrict__ cnt,
    const float* __restrict__ pocket,
    const float* __restrict__ pw1, const float* __restrict__ pb1,
    const float* __restrict__ pw2, const float* __restrict__ pb2,
    const float* __restrict__ cw1, const float* __restrict__ cb1,
    const float* __restrict__ cw2, const float* __restrict__ cb2,
    float* __restrict__ out) {
  int g = blockIdx.x, t = threadIdx.x;
  __shared__ float sp1[64], se[192], sc1[96], sred[128];
  if (t < 64) {
    float a = pb1[t];
#pragma unroll
    for (int k = 0; k < 28; ++k) a = fmaf(pocket[k], pw1[k * 64 + t], a);
    sp1[t] = fmaxf(a, 0.f);
  }
  float c = fmaxf(cnt[g], 1.f);
  if (t < 128) se[t] = sums[g * HID + t] / c;
  __syncthreads();
  if (t < 64) {
    float a = pb2[t];
    for (int k = 0; k < 64; ++k) a = fmaf(sp1[k], pw2[k * 64 + t], a);
    se[128 + t] = a;
  }
  __syncthreads();
  if (t < 96) {
    float a = cb1[t];
    for (int k = 0; k < 192; ++k) a = fmaf(se[k], cw1[k * 96 + t], a);
    sc1[t] = fmaxf(a, 0.f);
  }
  __syncthreads();
  sred[t] = (t < 96) ? sc1[t] * cw2[t] : 0.f;
  __syncthreads();
  for (int s = 64; s > 0; s >>= 1) {
    if (t < s) sred[t] += sred[t + s];
    __syncthreads();
  }
  if (t == 0) out[g] = sred[0] + cb2[0];
}

// ---------------- launcher ----------------
static inline size_t al256(size_t x) { return (x + 255) & ~(size_t)255; }

extern "C" void kernel_launch(void* const* d_in, const int* in_sizes, int n_in,
                              void* d_out, int out_size, void* d_ws, size_t ws_size,
                              hipStream_t stream) {
  const float* x      = (const float*)d_in[0];
  const int*   ei     = (const int*)  d_in[1];
  const int*   batch  = (const int*)  d_in[2];
  const float* pocket = (const float*)d_in[3];
  const float* W0 = (const float*)d_in[4];  const float* b0 = (const float*)d_in[5];
  const float* W1 = (const float*)d_in[6];  const float* b1 = (const float*)d_in[7];
  const float* W2 = (const float*)d_in[8];  const float* b2 = (const float*)d_in[9];
  const float* bn_g = (const float*)d_in[10];
  const float* bn_b = (const float*)d_in[11];
  const float* bn_m = (const float*)d_in[12];
  const float* bn_v = (const float*)d_in[13];
  const float* pw1 = (const float*)d_in[14]; const float* pb1 = (const float*)d_in[15];
  const float* pw2 = (const float*)d_in[16]; const float* pb2 = (const float*)d_in[17];
  const float* cw1 = (const float*)d_in[18]; const float* cb1 = (const float*)d_in[19];
  const float* cw2 = (const float*)d_in[20]; const float* cb2 = (const float*)d_in[21];
  float* out = (float*)d_out;

  const int N = N_NODES, E = N_EDGES;
  const size_t NH = (size_t)N * HID;

  char* ws = (char*)d_ws;
  size_t off = 0;
  float*  dinv = (float*)(ws + off);  off = al256(off + (size_t)N * 4);
  float*  hl   = (float*)(ws + off);  off = al256(off + NH * 4);
  float*  agg  = (float*)(ws + off);  off = al256(off + NH * 4);
  __bf16* hbf  = (__bf16*)(ws + off); off = al256(off + NH * 2);
  __bf16* W1bf = (__bf16*)(ws + off); off = al256(off + (size_t)HID * HID * 2);
  __bf16* W2bf = (__bf16*)(ws + off); off = al256(off + (size_t)HID * HID * 2);
  float*  sums = (float*)(ws + off);  off = al256(off + (size_t)(N_GRAPHS * HID + N_GRAPHS) * 4);
  float*  cnt  = sums + N_GRAPHS * HID;

  const int TB = 256;
  const int gN   = (N + TB - 1) / TB;
  const int gNH  = (int)((NH + TB - 1) / TB);
  const int gE   = (E + TB - 1) / TB;
  const int gE32 = (int)(((size_t)E * 32 + TB - 1) / TB);
  const int gW   = (HID * HID + TB - 1) / TB;
  const int gRow = (N + 127) / 128;   // 128-row blocks

  // degree -> rsqrt; zero pool accumulators
  zero_f<<<gN, TB, 0, stream>>>(dinv, N);
  zero_f<<<(N_GRAPHS * HID + N_GRAPHS + TB - 1) / TB, TB, 0, stream>>>(sums, N_GRAPHS * HID + N_GRAPHS);
  deg_count<<<gE, TB, 0, stream>>>(ei, dinv, E);
  make_dinv<<<gN, TB, 0, stream>>>(dinv, N);
  cvt_bf16_t<<<gW, TB, 0, stream>>>(W1, W1bf, HID * HID);
  cvt_bf16_t<<<gW, TB, 0, stream>>>(W2, W2bf, HID * HID);

  // layer 0 (K = 7, scalar GEMM)
  gemm_in7<<<gNH, TB, 0, stream>>>(x, W0, hl, N);
  init_agg<<<gNH, TB, 0, stream>>>(hl, dinv, b0, agg, N);
  edge_scatter<<<gE32, TB, 0, stream>>>(ei, hl, dinv, agg, E);
  bn_relu<<<gNH, TB, 0, stream>>>(agg, bn_g, bn_b, bn_m, bn_v, 0, hbf, N);

  // layer 1 (WMMA bf16)
  gemm128_wmma<<<gRow, TB, 0, stream>>>(hbf, W1bf, hl, N);
  init_agg<<<gNH, TB, 0, stream>>>(hl, dinv, b1, agg, N);
  edge_scatter<<<gE32, TB, 0, stream>>>(ei, hl, dinv, agg, E);
  bn_relu<<<gNH, TB, 0, stream>>>(agg, bn_g, bn_b, bn_m, bn_v, 1, hbf, N);

  // layer 2 (WMMA bf16)
  gemm128_wmma<<<gRow, TB, 0, stream>>>(hbf, W2bf, hl, N);
  init_agg<<<gNH, TB, 0, stream>>>(hl, dinv, b2, agg, N);
  edge_scatter<<<gE32, TB, 0, stream>>>(ei, hl, dinv, agg, E);
  bn_relu<<<gNH, TB, 0, stream>>>(agg, bn_g, bn_b, bn_m, bn_v, 2, hbf, N);

  // pool + head
  pool<<<gNH, TB, 0, stream>>>(hbf, batch, sums, cnt, N);
  head<<<N_GRAPHS, 128, 0, stream>>>(sums, cnt, pocket, pw1, pb1, pw2, pb2,
                                     cw1, cb1, cw2, cb2, out);
}
